// GraphSAGE_70317204570424
// MI455X (gfx1250) — matrix operand order repaired and checked by reading, not compiled, not run on hardware
//
#include <hip/hip_runtime.h>
#include <hip/hip_bf16.h>

// ---------------------------------------------------------------------------
// GraphSAGE (3x SAGEConv + mean-pool + linear) for MI455X / gfx1250.
//  - Edge aggregation: global_atomic_add_f32 (bandwidth path, L2-resident x)
//  - Dense [N x 256] @ [256 x 128] per layer: v_wmma_f32_16x16x32_bf16,
//    bf16 inputs / f32 accumulate, A staged in LDS, B from bf16 W^T in L2.
// ---------------------------------------------------------------------------

typedef __bf16  v16bf __attribute__((ext_vector_type(16)));
typedef float   v8f   __attribute__((ext_vector_type(8)));
typedef int     v4i   __attribute__((ext_vector_type(4)));

union FragAB { v16bf bf; v4i i[2]; };

__device__ __forceinline__ unsigned short f2bf(float f) {
    union { float f; unsigned u; } v; v.f = f;
    unsigned r = v.u + 0x7fffu + ((v.u >> 16) & 1u);   // round-to-nearest-even
    return (unsigned short)(r >> 16);
}

// ------------------------- utility kernels ---------------------------------

__global__ void zero_f32(float* __restrict__ p, long long n) {
    long long i = (long long)blockIdx.x * blockDim.x + threadIdx.x;
    long long stride = (long long)gridDim.x * blockDim.x;
    for (; i < n; i += stride) p[i] = 0.0f;
}

__global__ void degree_kernel(const int* __restrict__ ei, float* __restrict__ deg,
                              int n_edges) {
    int e = blockIdx.x * blockDim.x + threadIdx.x;
    if (e < n_edges) {
        int tgt = ei[n_edges + e];
        atomicAdd(&deg[tgt], 1.0f);
    }
}

// one wave per edge: 32 lanes x float4 = 128 floats
__global__ void scatter_kernel(const float* __restrict__ h, const int* __restrict__ ei,
                               float* __restrict__ msg, int n_edges) {
    int gid  = blockIdx.x * blockDim.x + threadIdx.x;
    int e    = gid >> 5;
    int lane = gid & 31;
    if (e >= n_edges) return;
    int src = ei[e];
    int tgt = ei[n_edges + e];
    float4 v = *reinterpret_cast<const float4*>(h + (long long)src * 128 + lane * 4);
    float* d = msg + (long long)tgt * 128 + lane * 4;
    atomicAdd(d + 0, v.x);
    atomicAdd(d + 1, v.y);
    atomicAdd(d + 2, v.z);
    atomicAdd(d + 3, v.w);
}

// W^T bf16 [128][256]:  Wt[n][k] = (k < 128 ? Wl[k][n] : Wr[k-128][n])
__global__ void cvt_weights(const float* __restrict__ Wl, const float* __restrict__ Wr,
                            unsigned short* __restrict__ Wt) {
    int i = blockIdx.x * blockDim.x + threadIdx.x;
    if (i >= 128 * 256) return;
    int n = i >> 8;
    int k = i & 255;
    float v = (k < 128) ? Wl[k * 128 + n] : Wr[(k - 128) * 128 + n];
    Wt[n * 256 + k] = f2bf(v);
}

// ------------------------- fused SAGE GEMM ---------------------------------
// out[i] = act( [msg[i]/max(deg,1) , hprev[i]] @ W^T  + bias )
// Block: 128 threads = 4 waves; 64 rows per block; wave -> 16 rows x 128 cols.

__global__ __launch_bounds__(128)
void sage_gemm(const float* __restrict__ msg, const float* __restrict__ deg,
               const float* __restrict__ hprev, const unsigned short* __restrict__ Wt,
               const float* __restrict__ bias, float* __restrict__ hout,
               int n_nodes, int relu) {
    __shared__ unsigned short sA[64 * 256];   // 32 KB bf16 A-tile [64 rows][K=256]
    __shared__ float sR[64];                  // 1/max(deg,1) per row

    const int tid     = threadIdx.x;
    const int rowBase = blockIdx.x * 64;

    if (tid < 64) {
        int gr = rowBase + tid;
        float d = (gr < n_nodes) ? deg[gr] : 1.0f;
        sR[tid] = 1.0f / fmaxf(d, 1.0f);
    }

    // warm the weight panel into near caches (global_prefetch_b8)
    __builtin_prefetch(Wt + (tid & 127) * 256, 0, 0);

    __syncthreads();

    // Stage A tile: 64 rows x 256 bf16, packed 2 cols per dword store.
    for (int i = tid; i < 64 * 128; i += 128) {
        int r  = i >> 7;            // 0..63
        int cp = (i & 127) * 2;     // 0..254 (even)
        int gr = rowBase + r;
        float v0 = 0.0f, v1 = 0.0f;
        if (gr < n_nodes) {
            if (cp < 128) {
                float s = sR[r];
                v0 = msg[(long long)gr * 128 + cp]     * s;
                v1 = msg[(long long)gr * 128 + cp + 1] * s;
            } else {
                v0 = hprev[(long long)gr * 128 + cp - 128];
                v1 = hprev[(long long)gr * 128 + cp - 127];
            }
        }
        unsigned pack = (unsigned)f2bf(v0) | ((unsigned)f2bf(v1) << 16);
        *reinterpret_cast<unsigned*>(&sA[r * 256 + cp]) = pack;
    }
    __syncthreads();

    const int wave   = tid >> 5;
    const int lane   = tid & 31;
    const int laneHi = lane >> 4;   // 0 | 1
    const int nl     = lane & 15;

    v8f acc[8];
    const v8f vzero = {0.f, 0.f, 0.f, 0.f, 0.f, 0.f, 0.f, 0.f};
    #pragma unroll
    for (int t = 0; t < 8; ++t) acc[t] = vzero;

    #pragma unroll
    for (int ks = 0; ks < 8; ++ks) {
        const int kbase = ks * 32;
        // A fragment (16x32 bf16): lane -> row = nl;
        //   K = kbase + laneHi*16 + 0..15 contiguous
        FragAB a;
        const unsigned short* ap = &sA[(wave * 16 + nl) * 256 + kbase + laneHi * 16];
        a.i[0] = *reinterpret_cast<const v4i*>(ap);
        a.i[1] = *reinterpret_cast<const v4i*>(ap + 8);

        #pragma unroll
        for (int t = 0; t < 8; ++t) {
            // B fragment (32x16 bf16): lane -> col n = t*16 + nl;
            //   K = kbase + laneHi*16 + 0..15 contiguous in W^T row
            FragAB b;
            const unsigned short* bp = Wt + (t * 16 + nl) * 256 + kbase + laneHi * 16;
            b.i[0] = *reinterpret_cast<const v4i*>(bp);
            b.i[1] = *reinterpret_cast<const v4i*>(bp + 8);

            acc[t] = __builtin_amdgcn_wmma_f32_16x16x32_bf16(
                false, a.bf, false, b.bf, (short)0, acc[t], false, false);
        }
    }

    // C/D layout: VGPR r, lane l -> M = r + 8*laneHi, N = nl
    #pragma unroll
    for (int t = 0; t < 8; ++t) {
        int col = t * 16 + nl;
        float bv = bias[col];
        #pragma unroll
        for (int r = 0; r < 8; ++r) {
            int grow = rowBase + wave * 16 + r + laneHi * 8;
            if (grow < n_nodes) {
                float v = acc[t][r] + bv;
                if (relu) v = fmaxf(v, 0.0f);
                hout[(long long)grow * 128 + col] = v;
            }
        }
    }
}

// ------------------------- pooling + head ----------------------------------

__global__ void pool_cnt_kernel(const int* __restrict__ batch, float* __restrict__ cnt,
                                int n_nodes) {
    int i = blockIdx.x * blockDim.x + threadIdx.x;
    if (i < n_nodes) atomicAdd(&cnt[batch[i]], 1.0f);
}

__global__ void pool_sum_kernel(const float* __restrict__ h, const int* __restrict__ batch,
                                float* __restrict__ pool, int n_nodes) {
    int gid  = blockIdx.x * blockDim.x + threadIdx.x;
    int node = gid >> 5;
    int lane = gid & 31;
    if (node >= n_nodes) return;
    int g = batch[node];
    float4 v = *reinterpret_cast<const float4*>(h + (long long)node * 128 + lane * 4);
    float* d = pool + g * 128 + lane * 4;
    atomicAdd(d + 0, v.x);
    atomicAdd(d + 1, v.y);
    atomicAdd(d + 2, v.z);
    atomicAdd(d + 3, v.w);
}

__global__ void head_kernel(const float* __restrict__ pool, const float* __restrict__ cnt,
                            const float* __restrict__ Wout, const float* __restrict__ bout,
                            float* __restrict__ out) {
    int g = blockIdx.x;
    int c = threadIdx.x;
    if (c >= 10) return;
    float inv = 1.0f / fmaxf(cnt[g], 1.0f);
    float s = bout[c];
    #pragma unroll 4
    for (int k = 0; k < 128; ++k)
        s += pool[g * 128 + k] * inv * Wout[k * 10 + c];
    out[g * 10 + c] = s;
}

// ------------------------- launch ------------------------------------------

static inline size_t align256(size_t x) { return (x + 255) & ~(size_t)255; }

extern "C" void kernel_launch(void* const* d_in, const int* in_sizes, int n_in,
                              void* d_out, int out_size, void* d_ws, size_t ws_size,
                              hipStream_t stream) {
    const float* x     = (const float*)d_in[0];
    const int*   ei    = (const int*)  d_in[1];   // [2][E]
    const int*   batch = (const int*)  d_in[2];
    const float* W1l = (const float*)d_in[3];
    const float* b1  = (const float*)d_in[4];
    const float* W1r = (const float*)d_in[5];
    const float* W2l = (const float*)d_in[6];
    const float* b2  = (const float*)d_in[7];
    const float* W2r = (const float*)d_in[8];
    const float* W3l = (const float*)d_in[9];
    const float* b3  = (const float*)d_in[10];
    const float* W3r = (const float*)d_in[11];
    const float* Wout = (const float*)d_in[12];
    const float* bout = (const float*)d_in[13];
    float* out = (float*)d_out;

    const int n_nodes = in_sizes[0] / 128;
    const int n_edges = in_sizes[1] / 2;
    const long long NF = (long long)n_nodes * 128;

    // workspace layout
    char* ws = (char*)d_ws;
    size_t off = 0;
    float* deg  = (float*)(ws + off); off += align256((size_t)n_nodes * 4);
    float* msg  = (float*)(ws + off); off += align256((size_t)NF * 4);
    float* hA   = (float*)(ws + off); off += align256((size_t)NF * 4);
    float* hB   = (float*)(ws + off); off += align256((size_t)NF * 4);
    float* pool = (float*)(ws + off); off += align256(64 * 128 * 4);
    float* cnt  = (float*)(ws + off); off += align256(64 * 4);
    unsigned short* Wt1 = (unsigned short*)(ws + off); off += align256(128 * 256 * 2);
    unsigned short* Wt2 = (unsigned short*)(ws + off); off += align256(128 * 256 * 2);
    unsigned short* Wt3 = (unsigned short*)(ws + off); off += align256(128 * 256 * 2);
    (void)ws_size;

    const int gemm_blocks = (n_nodes + 63) / 64;
    const int scat_blocks = (n_edges * 32 + 255) / 256;
    const int psum_blocks = (n_nodes * 32 + 255) / 256;

    // weights -> bf16 W^T
    cvt_weights<<<128, 256, 0, stream>>>(W1l, W1r, Wt1);
    cvt_weights<<<128, 256, 0, stream>>>(W2l, W2r, Wt2);
    cvt_weights<<<128, 256, 0, stream>>>(W3l, W3r, Wt3);

    // degrees
    zero_f32<<<512, 256, 0, stream>>>(deg, n_nodes);
    degree_kernel<<<(n_edges + 255) / 256, 256, 0, stream>>>(ei, deg, n_edges);

    // layer 1
    zero_f32<<<2048, 256, 0, stream>>>(msg, NF);
    scatter_kernel<<<scat_blocks, 256, 0, stream>>>(x, ei, msg, n_edges);
    sage_gemm<<<gemm_blocks, 128, 0, stream>>>(msg, deg, x, Wt1, b1, hA, n_nodes, 1);

    // layer 2
    zero_f32<<<2048, 256, 0, stream>>>(msg, NF);
    scatter_kernel<<<scat_blocks, 256, 0, stream>>>(hA, ei, msg, n_edges);
    sage_gemm<<<gemm_blocks, 128, 0, stream>>>(msg, deg, hA, Wt2, b2, hB, n_nodes, 1);

    // layer 3 (no relu)
    zero_f32<<<2048, 256, 0, stream>>>(msg, NF);
    scatter_kernel<<<scat_blocks, 256, 0, stream>>>(hB, ei, msg, n_edges);
    sage_gemm<<<gemm_blocks, 128, 0, stream>>>(msg, deg, hB, Wt3, b3, hA, n_nodes, 0);

    // global mean pool + head
    zero_f32<<<34, 256, 0, stream>>>(pool, 64 * 128);
    zero_f32<<<1, 64, 0, stream>>>(cnt, 64);
    pool_cnt_kernel<<<(n_nodes + 255) / 256, 256, 0, stream>>>(batch, cnt, n_nodes);
    pool_sum_kernel<<<psum_blocks, 256, 0, stream>>>(hA, batch, pool, n_nodes);
    head_kernel<<<64, 32, 0, stream>>>(pool, cnt, Wout, bout, out);
    (void)out_size; (void)n_in;
}
